// DAU_42949672960416
// MI455X (gfx1250) — matrix operand order repaired and checked by reading, not compiled, not run on hardware
//
#include <hip/hip_runtime.h>
#include <math.h>

typedef __attribute__((ext_vector_type(16))) _Float16 v16h;
typedef __attribute__((ext_vector_type(8)))  float    v8f;

#define NB   2
#define C_   256
#define H_   48
#define W_   48
#define HW   2304
#define HID  64
#define OH   96
#define OW   96
#define N1   9216
#define NG_  2
#define NH_  2
#define GC   32
#define RF_  11.0f

// ---------------- 1. channel LayerNorm -> f16 activations [pixel][256] ----
__global__ void ln_qk_prep(const float* __restrict__ x,
                           const float* __restrict__ g,
                           const float* __restrict__ bta,
                           _Float16* __restrict__ Xh) {
  int p = blockIdx.x;          // pixel 0..NB*HW-1
  int c = threadIdx.x;         // channel 0..255
  int bb = p / HW, hw = p % HW;
  float v = x[(bb * C_ + c) * HW + hw];
  __shared__ float s1[256], s2[256];
  s1[c] = v; s2[c] = v * v;
  __syncthreads();
  for (int o = 128; o > 0; o >>= 1) {
    if (c < o) { s1[c] += s1[c + o]; s2[c] += s2[c + o]; }
    __syncthreads();
  }
  float mu  = s1[0] * (1.0f / C_);
  float var = s2[0] * (1.0f / C_) - mu * mu;
  float xn  = (v - mu) * rsqrtf(var + 1e-5f) * g[c] + bta[c];
  Xh[p * C_ + c] = (_Float16)xn;
}

// ---------------- 2. pack Wq||Wk -> f16 [128][256] ------------------------
__global__ void cvt_w(const float* __restrict__ Wq, const float* __restrict__ Wk,
                      _Float16* __restrict__ Wh) {
  int i = blockIdx.x * 256 + threadIdx.x;
  if (i >= 128 * C_) return;
  int m = i / C_, kk = i % C_;
  float v = (m < HID) ? Wq[m * C_ + kk] : Wk[(m - HID) * C_ + kk];
  Wh[i] = (_Float16)v;
}

// ---------------- 3. WMMA GEMM: [128,256] x [256,4608] --------------------
// one wave per 16x16 tile; wave id in block = M-tile, blockIdx.x = N-tile
__global__ void gemm_qk(const _Float16* __restrict__ Wh,
                        const _Float16* __restrict__ Xh,
                        float* __restrict__ qo, float* __restrict__ ko) {
  int wave  = threadIdx.x >> 5;      // M tile 0..7
  int lane  = threadIdx.x & 31;
  int ntile = blockIdx.x;            // 0..287
  int lmod  = lane & 15;
  int lhalf = lane >> 4;
  v8f acc = {};
  const _Float16* arow = Wh + (size_t)(wave * 16 + lmod) * C_;
  const _Float16* brow = Xh + (size_t)(ntile * 16 + lmod) * C_;
  for (int k0 = 0; k0 < C_; k0 += 32) {
    v16h a, bf;
    // A 16x32 f16 layout: lanes 0-15 hold K{0..7,16..23}, lanes 16-31 K{8..15,24..31}
#pragma unroll
    for (int i = 0; i < 8; ++i) {
      a[i]     = arow[k0 + lhalf * 8 + i];
      a[i + 8] = arow[k0 + 16 + lhalf * 8 + i];
    }
    // B 32x16 f16 layout: lanes 0-15 hold K{0..7,16..23}, lanes 16-31 K{8..15,24..31}
#pragma unroll
    for (int i = 0; i < 16; ++i)
      bf[i] = brow[k0 + (i < 8 ? lhalf * 8 + i : 16 + lhalf * 8 + (i - 8))];
    acc = __builtin_amdgcn_wmma_f32_16x16x32_f16(
        /*neg_a=*/false, a, /*neg_b=*/false, bf,
        /*c_mod=*/(short)0, acc, /*reuse_a=*/false, /*reuse_b=*/false);
  }
  // D layout: lane<16 -> rows 0..7 ; lane>=16 -> rows 8..15 ; col = lane&15
  int col = ntile * 16 + lmod;
  int bb = col / HW, hw = col % HW;
#pragma unroll
  for (int j = 0; j < 8; ++j) {
    int m = wave * 16 + (lhalf ? 8 + j : j);
    float v = acc[j];
    if (m < HID) qo[(bb * HID + m) * HW + hw] = v;
    else         ko[(bb * HID + (m - HID)) * HW + hw] = v;
  }
}

// ---------------- 4. bilinear align_corners upsample q 48->96 -------------
__global__ void upsample_q(const float* __restrict__ q, float* __restrict__ qup) {
  int i = blockIdx.x * 256 + threadIdx.x;
  if (i >= NB * HID * OH * OW) return;
  int ox = i % OW; int t = i / OW;
  int oy = t % OH; t /= OH;
  int ch = t % HID; int bb = t / HID;
  const float s = (float)(H_ - 1) / (float)(OH - 1);
  float ys = oy * s, xs = ox * s;
  int y0 = min((int)floorf(ys), H_ - 2); float wy = ys - (float)y0;
  int x0 = min((int)floorf(xs), W_ - 2); float wx = xs - (float)x0;
  const float* base = q + (size_t)(bb * HID + ch) * HW;
  float v00 = base[y0 * W_ + x0],       v01 = base[y0 * W_ + x0 + 1];
  float v10 = base[(y0 + 1) * W_ + x0], v11 = base[(y0 + 1) * W_ + x0 + 1];
  qup[i] = (v00 * (1.f - wy) + v10 * wy) * (1.f - wx)
         + (v01 * (1.f - wy) + v11 * wy) * wx;
}

// ---------------- 5. depthwise 3x3 + channel-LN + GELU (wave per pos) -----
__global__ void offset_pre(const float* __restrict__ qup,
                           const float* __restrict__ dw_w,
                           const float* __restrict__ og,
                           const float* __restrict__ ob,
                           float* __restrict__ tbuf) {
  int wid  = (blockIdx.x * blockDim.x + threadIdx.x) >> 5;
  int lane = threadIdx.x & 31;                // channel 0..31
  if (wid >= NB * NG_ * N1) return;
  int pos = wid % N1; int bg = wid / N1;      // qup viewed as [4][32][96][96]
  int oy = pos / OW, ox = pos % OW;
  const float* base = qup + (size_t)(bg * GC + lane) * N1;
  float acc = 0.f;
#pragma unroll
  for (int dy = -1; dy <= 1; ++dy)
#pragma unroll
    for (int dx = -1; dx <= 1; ++dx) {
      int yy = oy + dy, xx = ox + dx;
      if (yy >= 0 && yy < OH && xx >= 0 && xx < OW)
        acc += base[yy * OW + xx] * dw_w[lane * 9 + (dy + 1) * 3 + (dx + 1)];
    }
  float s = acc, s2 = acc * acc;
  for (int m = 16; m > 0; m >>= 1) {
    s  += __shfl_xor(s,  m, 32);
    s2 += __shfl_xor(s2, m, 32);
  }
  float mu  = s * (1.f / GC);
  float var = s2 * (1.f / GC) - mu * mu;
  float xn  = (acc - mu) * rsqrtf(var + 1e-5f) * og[lane] + ob[lane];
  float ge  = 0.5f * xn * (1.0f + erff(xn * 0.70710678118654752f)); // exact GELU
  tbuf[(size_t)(bg * GC + lane) * N1 + pos] = ge;
}

// ---------------- 6. 3x3 conv 32->18 (+bias) ------------------------------
__global__ void offset_conv(const float* __restrict__ tbuf,
                            const float* __restrict__ ow_,
                            const float* __restrict__ obias,
                            float* __restrict__ pred) {
  int i = blockIdx.x * 256 + threadIdx.x;     // (bg,pos)
  if (i >= NB * NG_ * N1) return;
  int pos = i % N1, bg = i / N1;
  int oy = pos / OW, ox = pos % OW;
  float acc[18];
#pragma unroll
  for (int o = 0; o < 18; ++o) acc[o] = obias[o];
  for (int c = 0; c < GC; ++c) {
    const float* tb = tbuf + (size_t)(bg * GC + c) * N1;
#pragma unroll
    for (int dy = -1; dy <= 1; ++dy)
#pragma unroll
      for (int dx = -1; dx <= 1; ++dx) {
        int yy = oy + dy, xx = ox + dx;
        if (yy < 0 || yy >= OH || xx < 0 || xx >= OW) continue;
        float tv = tb[yy * OW + xx];
        int widx = c * 9 + (dy + 1) * 3 + (dx + 1);
#pragma unroll
        for (int o = 0; o < 18; ++o)
          acc[o] += tv * ow_[o * GC * 9 + widx];
      }
  }
  for (int o = 0; o < 18; ++o)
    pred[(size_t)(bg * 18 + o) * N1 + pos] = acc[o];
}

// ---------------- 7. deformable gather + attention (wave per pixel,head) --
__global__ void attn_kernel(const float* __restrict__ qup,
                            const float* __restrict__ kbuf,
                            const float* __restrict__ x,     // v = x
                            const float* __restrict__ pred,
                            const float* __restrict__ rpb,
                            float* __restrict__ out) {
  int wid  = (blockIdx.x * blockDim.x + threadIdx.x) >> 5;
  int lane = threadIdx.x & 31;               // channel within head (c2=32)
  if (wid >= NB * N1 * NH_) return;
  int h = wid & 1; int pix = wid >> 1;
  int bb = pix / N1; int p1 = pix % N1;
  int oy = p1 / OW, ox = p1 % OW;
  const float scale = 0.17677669529663687f;  // 32^-0.5
  float qv = qup[(size_t)(bb * HID + h * 32 + lane) * N1 + p1] * scale;
  const float* pr    = pred + (size_t)(bb * NG_ + h) * 18 * N1 + p1;
  const float* kbase = kbuf + (size_t)((bb * NG_ + h) * GC + lane) * HW;
  const float cs = (float)(H_ - 1) / (float)(OH - 1);
  int   y0s[9], x0s[9];
  float wys[9], wxs[9], logits[9];
  auto gatk = [&](int yy, int xx) -> float {
    if (yy < 0 || yy >= H_ || xx < 0 || xx >= W_) return 0.f;
    return kbase[yy * W_ + xx];
  };
#pragma unroll
  for (int t = 0; t < 9; ++t) {
    int ky = t / 3, kx = t % 3;
    float offy = tanhf(pr[(2 * t) * N1])     * RF_ + (float)(ky - 1);
    float offx = tanhf(pr[(2 * t + 1) * N1]) * RF_ + (float)(kx - 1);
    float iy = (offy + (float)oy) * cs;
    float ix = (offx + (float)ox) * cs;
    float yf = floorf(iy), xf = floorf(ix);
    int y0 = (int)yf, x0 = (int)xf;
    float wy = iy - yf, wx = ix - xf;
    y0s[t] = y0; x0s[t] = x0; wys[t] = wy; wxs[t] = wx;
    float kv = gatk(y0, x0)         * (1.f - wy) * (1.f - wx)
             + gatk(y0, x0 + 1)     * (1.f - wy) * wx
             + gatk(y0 + 1, x0)     * wy * (1.f - wx)
             + gatk(y0 + 1, x0 + 1) * wy * wx;
    kv += rpb[(h * 9 + t) * 32 + lane];
    float part = qv * kv;
    for (int m = 16; m > 0; m >>= 1) part += __shfl_xor(part, m, 32);
    logits[t] = part;
  }
  float mx = logits[0];
#pragma unroll
  for (int t = 1; t < 9; ++t) mx = fmaxf(mx, logits[t]);
  float se = 0.f;
#pragma unroll
  for (int t = 0; t < 9; ++t) { logits[t] = __expf(logits[t] - mx); se += logits[t]; }
  float inv = 1.f / se;

  // V: 128 channels per head -> 4 per lane (lane, lane+32, lane+64, lane+96)
  float a0 = 0.f, a1 = 0.f, a2 = 0.f, a3 = 0.f;
  const float* vb = x + (size_t)(bb * C_ + h * 128 + lane) * HW;
#pragma unroll
  for (int t = 0; t < 9; ++t) {
    float at = logits[t] * inv;
    int y0 = y0s[t], x0 = x0s[t];
    float wy = wys[t], wx = wxs[t];
    float cw[4] = { (1.f - wy) * (1.f - wx), (1.f - wy) * wx,
                    wy * (1.f - wx),         wy * wx };
    int cy[4] = { y0, y0, y0 + 1, y0 + 1 };
    int cx[4] = { x0, x0 + 1, x0, x0 + 1 };
#pragma unroll
    for (int cnr = 0; cnr < 4; ++cnr) {
      int yy = cy[cnr], xx = cx[cnr];
      if (yy < 0 || yy >= H_ || xx < 0 || xx >= W_) continue;
      int idx = yy * W_ + xx;
      float aw = at * cw[cnr];
      a0 += aw * vb[idx];
      a1 += aw * vb[32 * HW + idx];
      a2 += aw * vb[64 * HW + idx];
      a3 += aw * vb[96 * HW + idx];
    }
  }
  float* ob = out + (size_t)(bb * C_ + h * 128 + lane) * N1 + p1;
  ob[0]            = a0;
  ob[32 * (size_t)N1] = a1;
  ob[64 * (size_t)N1] = a2;
  ob[96 * (size_t)N1] = a3;
}

// --------------------------------------------------------------------------
extern "C" void kernel_launch(void* const* d_in, const int* in_sizes, int n_in,
                              void* d_out, int out_size, void* d_ws, size_t ws_size,
                              hipStream_t stream) {
  (void)in_sizes; (void)n_in; (void)out_size; (void)ws_size;
  const float* x       = (const float*)d_in[0];
  const float* ln_g    = (const float*)d_in[1];
  const float* ln_b    = (const float*)d_in[2];
  const float* Wq      = (const float*)d_in[3];
  const float* Wk      = (const float*)d_in[4];
  const float* dw_w    = (const float*)d_in[5];
  const float* off_g   = (const float*)d_in[6];
  const float* off_b   = (const float*)d_in[7];
  const float* off_w   = (const float*)d_in[8];
  const float* off_bias= (const float*)d_in[9];
  const float* rpb     = (const float*)d_in[10];
  float* out = (float*)d_out;

  char* ws = (char*)d_ws;
  size_t off = 0;
  auto alloc = [&](size_t bytes) -> void* {
    void* p = ws + off;
    off += (bytes + 255) & ~(size_t)255;
    return p;
  };
  _Float16* Wh  = (_Float16*)alloc((size_t)128 * C_ * sizeof(_Float16));
  _Float16* Xh  = (_Float16*)alloc((size_t)NB * HW * C_ * sizeof(_Float16));
  float*    qb  = (float*)alloc((size_t)NB * HID * HW * sizeof(float));
  float*    kb  = (float*)alloc((size_t)NB * HID * HW * sizeof(float));
  float*    qup = (float*)alloc((size_t)NB * HID * N1 * sizeof(float));
  float*    tb  = (float*)alloc((size_t)NB * NG_ * GC * N1 * sizeof(float));
  float*    pred= (float*)alloc((size_t)NB * NG_ * 18 * N1 * sizeof(float));

  cvt_w      <<<128,      256, 0, stream>>>(Wq, Wk, Wh);
  ln_qk_prep <<<NB * HW,  256, 0, stream>>>(x, ln_g, ln_b, Xh);
  gemm_qk    <<<288,      256, 0, stream>>>(Wh, Xh, qb, kb);     // 8 waves = 8 M-tiles
  upsample_q <<<(NB * HID * N1 + 255) / 256, 256, 0, stream>>>(qb, qup);
  offset_pre <<<(NB * NG_ * N1 * 32 + 255) / 256, 256, 0, stream>>>(qup, dw_w, off_g, off_b, tb);
  offset_conv<<<(NB * NG_ * N1 + 255) / 256, 256, 0, stream>>>(tb, off_w, off_bias, pred);
  attn_kernel<<<(NB * N1 * NH_ * 32 + 255) / 256, 256, 0, stream>>>(qup, kb, x, pred, rpb, out);
}
